// SynthRetroPretrainModel_59055800320566
// MI455X (gfx1250) — matrix operand order repaired and checked by reading, not compiled, not run on hardware
//
#include <hip/hip_runtime.h>
#include <hip/hip_bf16.h>
#include <math.h>

// Problem constants (from reference)
#define N_ATOMS  8192
#define N_BONDS  32768
#define N_GRAPHS 256
#define EMBED    32
#define HIDDEN   256
#define DEPTH    10
#define KSPLIT   16           // K-split of the big GEMM (deterministic partials, no atomics)

typedef __attribute__((ext_vector_type(16))) _Float16 v16h;
typedef __attribute__((ext_vector_type(8)))  _Float16 v8h;
typedef __attribute__((ext_vector_type(4)))  _Float16 v4h;
typedef __attribute__((ext_vector_type(8)))  float    v8f;

union V16U { v16h v; v8h h[2]; };

// ---------------------------------------------------------------------------
// Embedding gathers
// ---------------------------------------------------------------------------
__global__ void k_gather_atoms(const int* __restrict__ feat, const float* __restrict__ emb,
                               float* __restrict__ h, _Float16* __restrict__ hT) {
  int i = blockIdx.x * blockDim.x + threadIdx.x;     // 0 .. N_ATOMS*EMBED-1
  int a = i >> 5, e = i & 31;
  float v = emb[feat[a] * EMBED + e];
  h[i] = v;
  hT[(size_t)e * N_ATOMS + a] = (_Float16)v;         // transposed copy for WMMA B-fragments
}

__global__ void k_gather_bonds(const int* __restrict__ feat, const float* __restrict__ emb,
                               float* __restrict__ out) {
  int i = blockIdx.x * blockDim.x + threadIdx.x;     // 0 .. N_BONDS*EMBED-1
  int b = i >> 5, e = i & 31;
  out[i] = emb[feat[b] * EMBED + e];
}

// ---------------------------------------------------------------------------
// One-time: adjacency int32 -> f16 (exact for 0/1), row degrees, inv_deg mask.
// The 128 MiB f16 adjacency fits in the 192 MB L2, so layers 2..10 hit L2.
// ---------------------------------------------------------------------------
__global__ void k_prep_adj(const int* __restrict__ adj, _Float16* __restrict__ Ah,
                           float* __restrict__ inv_deg, float* __restrict__ has_nb) {
  int r = blockIdx.x;
  const int* row = adj + (size_t)r * N_ATOMS;
  _Float16* arow = Ah + (size_t)r * N_ATOMS;
  int cnt = 0;
  for (int base = 0; base < N_ATOMS; base += 1024) {
    int idx = base + threadIdx.x * 4;
    int4 v = *(const int4*)(row + idx);
    v4h o;
    o[0] = (v.x > 0) ? (_Float16)1.0f : (_Float16)0.0f;
    o[1] = (v.y > 0) ? (_Float16)1.0f : (_Float16)0.0f;
    o[2] = (v.z > 0) ? (_Float16)1.0f : (_Float16)0.0f;
    o[3] = (v.w > 0) ? (_Float16)1.0f : (_Float16)0.0f;
    *(v4h*)(arow + idx) = o;
    cnt += (v.x > 0) + (v.y > 0) + (v.z > 0) + (v.w > 0);
  }
  __shared__ int sred[256];
  sred[threadIdx.x] = cnt;
  __syncthreads();
  for (int s = 128; s > 0; s >>= 1) {
    if (threadIdx.x < s) sred[threadIdx.x] += sred[threadIdx.x + s];
    __syncthreads();
  }
  if (threadIdx.x == 0) {
    int d = sred[0];
    inv_deg[r] = (d > 0) ? 1.0f / (float)d : 0.0f;
    has_nb[r]  = (d > 0) ? 1.0f : 0.0f;
  }
}

// ---------------------------------------------------------------------------
// Big GEMM: partial[ks] = A_half[:, ksplit range] @ h via v_wmma_f32_16x16x32_f16.
// Each wave register-tiles a 32(M) x 32(N) output block: 2 A fragments x
// 2 B fragments -> 4 WMMAs per K-step, with B fragments shared across the two
// M-tiles (2 b128 loads per WMMA instead of 3). Block = 8 waves = 256 rows.
// grid = (8192/256 = 32 M-blocks, KSPLIT=16 K-splits).
// Fragment layouts per CDNA5 ISA 7.12.2:
//   A 16x32 f16 : lane L -> M = L%16 ; halves 0..7 = K (L/16)*8..+7, halves 8..15 = +16
//   B 32x16 f16 : lane L -> N = L%16 ; halves 0..7 = K (L/16)*8..+7, halves 8..15 = +16
// ---------------------------------------------------------------------------
__global__ void k_spmm_wmma(const _Float16* __restrict__ Ah, const _Float16* __restrict__ hT,
                            float* __restrict__ partial) {
  const int lane = threadIdx.x & 31;
  const int wave = threadIdx.x >> 5;
  const int m0   = (blockIdx.x * 8 + wave) * 32;   // 32 rows per wave
  const int lh   = lane >> 4;                      // which half of the wave
  const int ln   = lane & 15;
  const int k_begin = blockIdx.y * (N_ATOMS / KSPLIT);
  const int k_end   = k_begin + (N_ATOMS / KSPLIT);

  v8f acc00 = {}; v8f acc01 = {}; v8f acc10 = {}; v8f acc11 = {};
  const _Float16* arow0 = Ah + (size_t)(m0 + ln) * N_ATOMS;
  const _Float16* arow1 = Ah + (size_t)(m0 + 16 + ln) * N_ATOMS;
  const _Float16* b0row = hT + (size_t)ln * N_ATOMS;
  const _Float16* b1row = hT + (size_t)(ln + 16) * N_ATOMS;

  for (int k0 = k_begin; k0 < k_end; k0 += 32) {
    V16U a0, a1, b0, b1;
    const int ka = k0 + lh * 8;
    a0.h[0] = *(const v8h*)(arow0 + ka);
    a0.h[1] = *(const v8h*)(arow0 + ka + 16);
    a1.h[0] = *(const v8h*)(arow1 + ka);
    a1.h[1] = *(const v8h*)(arow1 + ka + 16);
    const int kb = k0 + lh * 8;
    b0.h[0] = *(const v8h*)(b0row + kb);
    b0.h[1] = *(const v8h*)(b0row + kb + 16);
    b1.h[0] = *(const v8h*)(b1row + kb);
    b1.h[1] = *(const v8h*)(b1row + kb + 16);
    acc00 = __builtin_amdgcn_wmma_f32_16x16x32_f16(false, a0.v, false, b0.v,
                                                   (short)0, acc00, false, false);
    acc01 = __builtin_amdgcn_wmma_f32_16x16x32_f16(false, a0.v, false, b1.v,
                                                   (short)0, acc01, false, false);
    acc10 = __builtin_amdgcn_wmma_f32_16x16x32_f16(false, a1.v, false, b0.v,
                                                   (short)0, acc10, false, false);
    acc11 = __builtin_amdgcn_wmma_f32_16x16x32_f16(false, a1.v, false, b1.v,
                                                   (short)0, acc11, false, false);
  }

  // C/D layout: VGPR v -> row tile_m0 + v + 8*(lane/16), col = lane%16
  float* out = partial + (size_t)blockIdx.y * N_ATOMS * EMBED;
  #pragma unroll
  for (int v = 0; v < 8; ++v) {
    int row0 = m0 + v + lh * 8;
    int row1 = row0 + 16;
    out[(size_t)row0 * EMBED + ln]      = acc00[v];
    out[(size_t)row0 * EMBED + ln + 16] = acc01[v];
    out[(size_t)row1 * EMBED + ln]      = acc10[v];
    out[(size_t)row1 * EMBED + ln + 16] = acc11[v];
  }
}

// ---------------------------------------------------------------------------
// msg = h @ W1^T + neigh_mean @ W2^T + b, masked by has_nb.
// Fuses K-split reduction + inv_deg scaling. Block = one atom, 256 threads.
// ---------------------------------------------------------------------------
__global__ void k_msg(const float* __restrict__ h, const float* __restrict__ partial,
                      const float* __restrict__ inv_deg, const float* __restrict__ has_nb,
                      const float* __restrict__ W, const float* __restrict__ b,
                      float* __restrict__ msg) {
  int a = blockIdx.x;
  int t = threadIdx.x;
  __shared__ float sh[64];                      // [0:32)=h row, [32:64)=neigh_mean row
  if (t < 32) {
    sh[t] = h[a * EMBED + t];
  } else if (t < 64) {
    int e = t - 32;
    float s = 0.f;
    #pragma unroll
    for (int ks = 0; ks < KSPLIT; ++ks)
      s += partial[(size_t)ks * N_ATOMS * EMBED + a * EMBED + e];
    sh[32 + e] = s * inv_deg[a];
  }
  __syncthreads();
  float mask = has_nb[a];
  const float* wrow = W + t * (2 * EMBED);      // W row j: [0:32)=W1, [32:64)=W2
  float acc = b[t];
  #pragma unroll
  for (int e = 0; e < EMBED; ++e)
    acc += sh[e] * wrow[e] + sh[32 + e] * wrow[32 + e];
  msg[(size_t)a * HIDDEN + t] = acc * mask;
}

// ---------------------------------------------------------------------------
// GRU cell: gi = msg@wih^T+bih, gh = h@whh^T+bhh, gates, h_new. Block = atom.
// Writes both h (f32) and the transposed f16 copy for the next layer's WMMA.
// ---------------------------------------------------------------------------
__global__ void k_gru(const float* __restrict__ msg, const float* __restrict__ wih,
                      const float* __restrict__ whh, const float* __restrict__ bih,
                      const float* __restrict__ bhh,
                      float* __restrict__ h, _Float16* __restrict__ hT) {
  int a = blockIdx.x, t = threadIdx.x;           // blockDim = 128
  __shared__ float smsg[HIDDEN];
  __shared__ float shh[EMBED];
  __shared__ float sgi[96], sgh[96];
  smsg[t]       = msg[(size_t)a * HIDDEN + t];
  smsg[t + 128] = msg[(size_t)a * HIDDEN + t + 128];
  if (t < EMBED) shh[t] = h[a * EMBED + t];
  __syncthreads();
  if (t < 96) {
    const float* wi = wih + t * HIDDEN;
    float acc = bih[t];
    for (int k = 0; k < HIDDEN; ++k) acc += smsg[k] * wi[k];
    sgi[t] = acc;
    const float* wh = whh + t * EMBED;
    float acc2 = bhh[t];
    #pragma unroll
    for (int e = 0; e < EMBED; ++e) acc2 += shh[e] * wh[e];
    sgh[t] = acc2;
  }
  __syncthreads();
  if (t < EMBED) {
    float r  = 1.f / (1.f + expf(-(sgi[t]      + sgh[t])));
    float z  = 1.f / (1.f + expf(-(sgi[32 + t] + sgh[32 + t])));
    float n  = tanhf(sgi[64 + t] + r * sgh[64 + t]);
    float hn = (1.f - z) * n + z * shh[t];
    h[a * EMBED + t] = hn;
    hT[(size_t)t * N_ATOMS + a] = (_Float16)hn;
  }
}

// ---------------------------------------------------------------------------
// Segment-mean pooling + graph_emb = mean @ pool_W^T + pool_b.
// Block = one graph; deterministic fixed-order tree reductions (no fp atomics).
// ---------------------------------------------------------------------------
__global__ void k_pool(const float* __restrict__ h, const int* __restrict__ batch,
                       const float* __restrict__ pW, const float* __restrict__ pb,
                       float* __restrict__ out_graph) {
  int g = blockIdx.x, t = threadIdx.x;           // blockDim = 256
  float acc[EMBED];
  #pragma unroll
  for (int e = 0; e < EMBED; ++e) acc[e] = 0.f;
  float cnt = 0.f;
  for (int a = t; a < N_ATOMS; a += 256) {
    if (batch[a] == g) {
      cnt += 1.f;
      const float* hr = h + a * EMBED;
      #pragma unroll
      for (int e = 0; e < EMBED; ++e) acc[e] += hr[e];
    }
  }
  __shared__ float red[256];
  __shared__ float mean[EMBED + 1];
  red[t] = cnt; __syncthreads();
  for (int s = 128; s > 0; s >>= 1) { if (t < s) red[t] += red[t + s]; __syncthreads(); }
  if (t == 0) mean[EMBED] = red[0];
  __syncthreads();
  for (int e = 0; e < EMBED; ++e) {
    red[t] = acc[e]; __syncthreads();
    for (int s = 128; s > 0; s >>= 1) { if (t < s) red[t] += red[t + s]; __syncthreads(); }
    if (t == 0) {
      float c = mean[EMBED];
      mean[e] = (c > 0.f) ? red[0] / c : 0.f;
    }
    __syncthreads();
  }
  float ge = pb[t];
  #pragma unroll
  for (int e = 0; e < EMBED; ++e) ge += mean[e] * pW[t * EMBED + e];
  out_graph[(size_t)g * HIDDEN + t] = ge;
}

// ---------------------------------------------------------------------------
extern "C" void kernel_launch(void* const* d_in, const int* in_sizes, int n_in,
                              void* d_out, int out_size, void* d_ws, size_t ws_size,
                              hipStream_t stream) {
  const int*   atom_features = (const int*)d_in[0];
  const int*   bond_features = (const int*)d_in[1];
  const int*   adj           = (const int*)d_in[2];
  const int*   batch         = (const int*)d_in[3];
  const float* atom_emb      = (const float*)d_in[4];
  const float* bond_emb      = (const float*)d_in[5];
  const float* msg_W         = (const float*)d_in[6];
  const float* msg_b         = (const float*)d_in[7];
  const float* gru_Wih       = (const float*)d_in[8];
  const float* gru_Whh       = (const float*)d_in[9];
  const float* gru_bih       = (const float*)d_in[10];
  const float* gru_bhh       = (const float*)d_in[11];
  const float* pool_W        = (const float*)d_in[12];
  const float* pool_b        = (const float*)d_in[13];

  // Workspace layout (~161 MB): f16 adjacency first so it stays L2-resident.
  char* ws = (char*)d_ws;
  size_t off = 0;
  _Float16* A_half = (_Float16*)(ws + off); off += (size_t)N_ATOMS * N_ATOMS * sizeof(_Float16);
  float*    h      = (float*)   (ws + off); off += (size_t)N_ATOMS * EMBED * sizeof(float);
  _Float16* hT     = (_Float16*)(ws + off); off += (size_t)N_ATOMS * EMBED * sizeof(_Float16);
  float*    invd   = (float*)   (ws + off); off += (size_t)N_ATOMS * sizeof(float);
  float*    hasnb  = (float*)   (ws + off); off += (size_t)N_ATOMS * sizeof(float);
  float*    part   = (float*)   (ws + off); off += (size_t)KSPLIT * N_ATOMS * EMBED * sizeof(float);
  float*    msg    = (float*)   (ws + off); off += (size_t)N_ATOMS * HIDDEN * sizeof(float);
  (void)ws_size; (void)in_sizes; (void)n_in; (void)out_size;

  float* out_H     = (float*)d_out;                       // 8192*32
  float* out_bond  = out_H + (size_t)N_ATOMS * EMBED;     // 32768*32
  float* out_graph = out_bond + (size_t)N_BONDS * EMBED;  // 256*256

  k_gather_atoms<<<(N_ATOMS * EMBED) / 256, 256, 0, stream>>>(atom_features, atom_emb, h, hT);
  k_gather_bonds<<<(N_BONDS * EMBED) / 256, 256, 0, stream>>>(bond_features, bond_emb, out_bond);
  k_prep_adj<<<N_ATOMS, 256, 0, stream>>>(adj, A_half, invd, hasnb);

  for (int d = 0; d < DEPTH; ++d) {
    k_spmm_wmma<<<dim3(N_ATOMS / 256, KSPLIT), 256, 0, stream>>>(A_half, hT, part);
    k_msg<<<N_ATOMS, 256, 0, stream>>>(h, part, invd, hasnb,
                                       msg_W + (size_t)d * HIDDEN * 2 * EMBED,
                                       msg_b + (size_t)d * HIDDEN, msg);
    k_gru<<<N_ATOMS, 128, 0, stream>>>(msg,
                                       gru_Wih + (size_t)d * 96 * HIDDEN,
                                       gru_Whh + (size_t)d * 96 * EMBED,
                                       gru_bih + (size_t)d * 96,
                                       gru_bhh + (size_t)d * 96, h, hT);
  }

  k_pool<<<N_GRAPHS, 256, 0, stream>>>(h, batch, pool_W, pool_b, out_graph);
  hipMemcpyAsync(out_H, h, (size_t)N_ATOMS * EMBED * sizeof(float),
                 hipMemcpyDeviceToDevice, stream);
}
